// AlexnetCA_27711128993804
// MI455X (gfx1250) — matrix operand order repaired and checked by reading, not compiled, not run on hardware
//
#include <hip/hip_runtime.h>

typedef __attribute__((ext_vector_type(16))) _Float16 v16h;
typedef __attribute__((ext_vector_type(8)))  _Float16 v8h;
typedef __attribute__((ext_vector_type(8)))  float    v8f;

#if defined(__has_builtin)
#if __has_builtin(__builtin_amdgcn_global_load_async_to_lds_b128) && \
    __has_builtin(__builtin_amdgcn_s_wait_asynccnt)
#define AC5_ASYNC 1
#endif
#endif
#ifndef AC5_ASYNC
#define AC5_ASYNC 0
#endif

#if AC5_ASYNC
// Builtin prototype (from clang diagnostic): (v4i AS1*, v4i AS3*, imm, imm)
typedef __attribute__((ext_vector_type(4))) int ac5_v4i;
typedef __attribute__((address_space(1))) ac5_v4i* ac5_gp;   // global src
typedef __attribute__((address_space(3))) ac5_v4i* ac5_lp;   // LDS dst
#endif

// ---------------------------------------------------------------------------
// im2col: fp32 NCHW input -> fp16 col [M = B*Ho*Wo, Kp] row-major, K padded->Kp
// ---------------------------------------------------------------------------
__global__ void AlexCA_im2col(const float* __restrict__ in, _Float16* __restrict__ col,
                              int Cin, int H, int W, int Ho, int Wo,
                              int ksz, int stride, int pad, int K, int Kp,
                              unsigned long long total) {
  unsigned long long idx = (unsigned long long)blockIdx.x * blockDim.x + threadIdx.x;
  if (idx >= total) return;
  int k = (int)(idx % Kp);
  unsigned long long m = idx / Kp;
  float val = 0.0f;
  if (k < K) {
    int kk = ksz * ksz;
    int c = k / kk;
    int rs = k - c * kk;
    int r = rs / ksz, s = rs - (rs / ksz) * ksz;
    int ow = (int)(m % Wo);
    int oh = (int)((m / Wo) % Ho);
    int b  = (int)(m / ((unsigned long long)Wo * Ho));
    int ih = oh * stride - pad + r;
    int iw = ow * stride - pad + s;
    if (ih >= 0 && ih < H && iw >= 0 && iw < W)
      val = in[(((size_t)b * Cin + c) * H + ih) * W + iw];
  }
  col[idx] = (_Float16)val;
}

// ---------------------------------------------------------------------------
// pack weights: fp32 w[N][K] (row-major) -> fp16 B [Kp][Np] row-major (transposed)
// ---------------------------------------------------------------------------
__global__ void AlexCA_packw(const float* __restrict__ w, _Float16* __restrict__ Bm,
                             int N, int K, int Np, unsigned long long total) {
  unsigned long long idx = (unsigned long long)blockIdx.x * blockDim.x + threadIdx.x;
  if (idx >= total) return;
  int n = (int)(idx % Np);
  unsigned long long k = idx / Np;
  float val = 0.0f;
  if (n < N && k < (unsigned long long)K) val = w[(size_t)n * K + k];
  Bm[idx] = (_Float16)val;
}

// ---------------------------------------------------------------------------
// WMMA GEMM core. A[M,Kp] fp16 row-major (direct global), B[Kp,Np] fp16
// row-major staged through double-buffered LDS (64K x 64N slab) with async
// copies. 8 waves: block tile 128Mx64N, wave tile 32x32 (2x2 wmma).
// SPLIT=false: C = relu(A*B + bias) stored NCHW via HW remap.
// SPLIT=true : partial C[z][M][Np] fp32, K-chunk = kLen per blockIdx.z.
// ---------------------------------------------------------------------------
template <bool SPLIT>
__global__ __launch_bounds__(256)
void AlexCA_gemm_core(const _Float16* __restrict__ A, const _Float16* __restrict__ Bm,
                      const float* __restrict__ bias, float* __restrict__ C,
                      int M, int Nout, int Kp, int Np, int kLen, int HW, int relu) {
  __shared__ alignas(64) _Float16 bsh[2][64 * 64];

  const int tid  = threadIdx.x;
  const int lane = tid & 31;
  const int wave = tid >> 5;
  const int wm = wave >> 1;                 // 0..3
  const int wn = wave & 1;                  // 0..1
  const int m_base = blockIdx.y * 128 + wm * 32;
  const int n_blk  = blockIdx.x * 64;
  const int kBase  = SPLIT ? blockIdx.z * kLen : 0;

  // A fragment rows (clamped; out-of-range rows discarded at store)
  int r0 = m_base + (lane & 15);
  int r1 = r0 + 16;
  if (r0 >= M) r0 = M - 1;
  if (r1 >= M) r1 = M - 1;
  const int ak = (lane >> 4) * 16;
  const _Float16* a0p = A + (size_t)r0 * Kp + ak;
  const _Float16* a1p = A + (size_t)r1 * Kp + ak;

  // B staging: 512 chunks of 8 halves (16B) per 64x64 slab; each of the 256
  // threads moves 2 chunks. chunk c: k_local = c>>3, n_off = (c&7)*8.
  const _Float16* gstage = Bm + (size_t)(kBase + (tid >> 3)) * Np + n_blk + (tid & 7) * 8;
  _Float16* lstage = &bsh[0][(tid >> 3) * 64 + (tid & 7) * 8];

  auto stage = [&](int buf, int kOff) {
    const _Float16* ga = gstage + (size_t)kOff * Np;
    const _Float16* gb = ga + (size_t)32 * Np;
    _Float16* la = lstage + buf * (64 * 64);
    _Float16* lb = la + 32 * 64;
#if AC5_ASYNC
    __builtin_amdgcn_global_load_async_to_lds_b128((ac5_gp)ga, (ac5_lp)la, 0, 0);
    __builtin_amdgcn_global_load_async_to_lds_b128((ac5_gp)gb, (ac5_lp)lb, 0, 0);
#else
    *(v8h*)la = *(const v8h*)ga;
    *(v8h*)lb = *(const v8h*)gb;
#endif
  };

  v8f acc00 = {}, acc01 = {}, acc10 = {}, acc11 = {};

  const int nSteps = kLen >> 6;             // kLen multiple of 64
  stage(0, 0);
  for (int s = 0; s < nSteps; ++s) {
    if (s + 1 < nSteps) {
      stage((s + 1) & 1, (s + 1) * 64);
#if AC5_ASYNC
      __builtin_amdgcn_s_wait_asynccnt(2);  // our 2 ops for slab s are done (in-order)
#endif
    } else {
#if AC5_ASYNC
      __builtin_amdgcn_s_wait_asynccnt(0);
#endif
    }
    __syncthreads();                        // slab s visible to all waves

    const _Float16* bb = &bsh[s & 1][0];
#pragma unroll
    for (int kk = 0; kk < 64; kk += 32) {
      const int kg = kBase + s * 64 + kk;
      v16h a0 = *(const v16h*)(a0p + kg);
      v16h a1 = *(const v16h*)(a1p + kg);
      const _Float16* bl = bb + (size_t)(kk + (lane >> 4) * 16) * 64 + wn * 32 + (lane & 15);
      v16h b0, b1;
#pragma unroll
      for (int i = 0; i < 16; ++i) {
        b0[i] = bl[i * 64];
        b1[i] = bl[i * 64 + 16];
      }
      acc00 = __builtin_amdgcn_wmma_f32_16x16x32_f16(false, a0, false, b0, (short)0, acc00, false, false);
      acc01 = __builtin_amdgcn_wmma_f32_16x16x32_f16(false, a0, false, b1, (short)0, acc01, false, false);
      acc10 = __builtin_amdgcn_wmma_f32_16x16x32_f16(false, a1, false, b0, (short)0, acc10, false, false);
      acc11 = __builtin_amdgcn_wmma_f32_16x16x32_f16(false, a1, false, b1, (short)0, acc11, false, false);
    }
    __syncthreads();                        // safe to overwrite slab s buffer
  }

  // C/D layout: VGPR v, lanes 0-15 -> (m=v, n=lane); lanes 16-31 -> (m=v+8, n=lane-16)
  const int colHalf = lane & 15;
  const int rowHalf = (lane >> 4) * 8;
#pragma unroll
  for (int mt = 0; mt < 2; ++mt) {
#pragma unroll
    for (int nt = 0; nt < 2; ++nt) {
      v8f acc = (mt == 0) ? (nt == 0 ? acc00 : acc01) : (nt == 0 ? acc10 : acc11);
      int col = n_blk + wn * 32 + nt * 16 + colHalf;
      if (SPLIT) {
#pragma unroll
        for (int v = 0; v < 8; ++v) {
          int row = m_base + mt * 16 + rowHalf + v;
          if (row < M)
            C[((size_t)blockIdx.z * M + row) * Np + col] = acc[v];
        }
      } else if (col < Nout) {
        float bv = bias[col];
#pragma unroll
        for (int v = 0; v < 8; ++v) {
          int row = m_base + mt * 16 + rowHalf + v;
          if (row < M) {
            float val = acc[v] + bv;
            if (relu) val = fmaxf(val, 0.0f);
            int b = row / HW;
            int p = row - b * HW;
            C[((size_t)b * Nout + col) * HW + p] = val;
          }
        }
      }
    }
  }
}

// reduce split-K partials: out[b,n] = bc[n] + sum_z part[z][b][n]
__global__ void AlexCA_fc_reduce(const float* __restrict__ part, const float* __restrict__ bias,
                                 float* __restrict__ out, int nsplit, int Np,
                                 unsigned long long total) {
  unsigned long long idx = (unsigned long long)blockIdx.x * blockDim.x + threadIdx.x;
  if (idx >= total) return;
  int n = (int)(idx % 1000);
  int b = (int)(idx / 1000);
  float s = bias[n];
  for (int z = 0; z < nsplit; ++z)
    s += part[((size_t)z * 64 + b) * Np + n];
  out[idx] = s;
}

// ---------------------------------------------------------------------------
// maxpool 3x3 stride 2 VALID on fp32 NCHW (flattened channels)
// ---------------------------------------------------------------------------
__global__ void AlexCA_maxpool(const float* __restrict__ in, float* __restrict__ out,
                               int CT, int H, int W, int Ho, int Wo,
                               unsigned long long total) {
  unsigned long long idx = (unsigned long long)blockIdx.x * blockDim.x + threadIdx.x;
  if (idx >= total) return;
  int ow = (int)(idx % Wo);
  int oh = (int)((idx / Wo) % Ho);
  int c  = (int)(idx / ((unsigned long long)Wo * Ho));
  const float* p = in + ((size_t)c * H + oh * 2) * W + ow * 2;
  float m = p[0];
#pragma unroll
  for (int dy = 0; dy < 3; ++dy)
#pragma unroll
    for (int dx = 0; dx < 3; ++dx)
      m = fmaxf(m, p[(size_t)dy * W + dx]);
  out[idx] = m;
}

// ---------------------------------------------------------------------------
// per-sample cmap stats
// ---------------------------------------------------------------------------
__global__ __launch_bounds__(256)
void AlexCA_stats(const float* __restrict__ pool3, float* __restrict__ mv,
                  float* __restrict__ q) {
  __shared__ float red[256];
  const int b = blockIdx.x;
  const int tid = threadIdx.x;
  const float* h = pool3 + (size_t)b * 9216;
  float m = -3.402823466e38f;
  for (int i = tid; i < 9216; i += 256) m = fmaxf(m, h[i]);
  red[tid] = m;
  __syncthreads();
  for (int s = 128; s > 0; s >>= 1) {
    if (tid < s) red[tid] = fmaxf(red[tid], red[tid + s]);
    __syncthreads();
  }
  const float maxf = red[0];
  const float inv = (maxf != 0.0f) ? 1.0f / maxf : 0.0f;
  const float* hk = h + tid * 36;
  float sum = 0.0f, mx = -3.402823466e38f;
#pragma unroll
  for (int j = 0; j < 36; ++j) {
    float v = hk[j] * inv;
    sum += v;
    mx = fmaxf(mx, v);
  }
  mv[b * 256 + tid] = mx;
  q[b * 256 + tid] = (sum != 0.0f) ? mx / sum : 0.0f;
}

// feat[b, 0:9216] = pool3[b] (fp16)
__global__ void AlexCA_featA(const float* __restrict__ pool3, _Float16* __restrict__ feat,
                             unsigned long long total) {
  unsigned long long idx = (unsigned long long)blockIdx.x * blockDim.x + threadIdx.x;
  if (idx >= total) return;
  int m = (int)(idx % 9216);
  int b = (int)(idx / 9216);
  feat[(size_t)b * 74752 + m] = (_Float16)pool3[idx];
}

// feat[b, 9216 + i*256 + j] = mv[b,i] * q[b,j]
__global__ void AlexCA_featB(const float* __restrict__ mv, const float* __restrict__ q,
                             _Float16* __restrict__ feat, unsigned long long total) {
  unsigned long long idx = (unsigned long long)blockIdx.x * blockDim.x + threadIdx.x;
  if (idx >= total) return;
  int r = (int)(idx & 65535);
  int b = (int)(idx >> 16);
  int i = r >> 8;
  int j = r & 255;
  feat[(size_t)b * 74752 + 9216 + r] = (_Float16)(mv[b * 256 + i] * q[b * 256 + j]);
}

// ---------------------------------------------------------------------------
// host-side orchestration
// ---------------------------------------------------------------------------
extern "C" void kernel_launch(void* const* d_in, const int* in_sizes, int n_in,
                              void* d_out, int out_size, void* d_ws, size_t ws_size,
                              hipStream_t stream) {
  const float* x  = (const float*)d_in[0];
  const float* w1 = (const float*)d_in[1];
  const float* b1 = (const float*)d_in[2];
  const float* w2 = (const float*)d_in[3];
  const float* b2 = (const float*)d_in[4];
  const float* w3 = (const float*)d_in[5];
  const float* b3 = (const float*)d_in[6];
  const float* w4 = (const float*)d_in[7];
  const float* b4 = (const float*)d_in[8];
  const float* w5 = (const float*)d_in[9];
  const float* b5 = (const float*)d_in[10];
  const float* wc = (const float*)d_in[11];
  const float* bc = (const float*)d_in[12];
  float* out = (float*)d_out;

  constexpr int NSPLIT = 73;      // FC split-K: 73 chunks of K=1024 (73*1024 = 74752)
  constexpr int KCH    = 1024;

  char* ws = (char*)d_ws;
  size_t off = 0;
  auto take = [&](size_t bytes) -> void* {
    void* p = ws + off;
    off += (bytes + 255) & ~(size_t)255;
    return p;
  };
  _Float16* col = (_Float16*)take(149299200);   // max im2col (conv2: 46656*1600*2)
  _Float16* Bm  = (_Float16*)take(153092096);   // max packed weights (wc: 74752*1024*2)
  float* act1  = (float*)take((size_t)64 * 64 * 55 * 55 * 4);
  float* pool1 = (float*)take((size_t)64 * 64 * 27 * 27 * 4);
  float* act2  = (float*)take((size_t)64 * 192 * 27 * 27 * 4);
  float* pool2 = (float*)take((size_t)64 * 192 * 13 * 13 * 4);
  float* act3  = (float*)take((size_t)64 * 384 * 13 * 13 * 4);
  float* act4  = (float*)take((size_t)64 * 256 * 13 * 13 * 4);
  float* act5  = (float*)take((size_t)64 * 256 * 13 * 13 * 4);
  float* pool3 = (float*)take((size_t)64 * 256 * 6 * 6 * 4);
  float* mv    = (float*)take((size_t)64 * 256 * 4);
  float* qv    = (float*)take((size_t)64 * 256 * 4);
  _Float16* feat = (_Float16*)take((size_t)64 * 74752 * 2);
  float* part  = (float*)take((size_t)NSPLIT * 64 * 1024 * 4);

  auto g1 = [](unsigned long long n) { return dim3((unsigned)((n + 255) / 256)); };
  auto cdiv = [](int a, int b) { return (a + b - 1) / b; };

  // ---- conv1: 3x224x224 -> 64x55x55, k11 s4 p2, K=363->384 ----
  {
    const int M = 64 * 55 * 55, K = 363, Kp = 384, N = 64, Np = 64;
    unsigned long long tc = (unsigned long long)M * Kp;
    AlexCA_im2col<<<g1(tc), 256, 0, stream>>>(x, col, 3, 224, 224, 55, 55, 11, 4, 2, K, Kp, tc);
    unsigned long long tw = (unsigned long long)Kp * Np;
    AlexCA_packw<<<g1(tw), 256, 0, stream>>>(w1, Bm, N, K, Np, tw);
    AlexCA_gemm_core<false><<<dim3(Np / 64, cdiv(M, 128)), 256, 0, stream>>>(
        col, Bm, b1, act1, M, N, Kp, Np, Kp, 55 * 55, 1);
    unsigned long long tp = (unsigned long long)64 * 64 * 27 * 27;
    AlexCA_maxpool<<<g1(tp), 256, 0, stream>>>(act1, pool1, 64 * 64, 55, 55, 27, 27, tp);
  }
  // ---- conv2: 64x27x27 -> 192x27x27, k5 s1 p2, K=1600 ----
  {
    const int M = 64 * 27 * 27, K = 1600, Kp = 1600, N = 192, Np = 192;
    unsigned long long tc = (unsigned long long)M * Kp;
    AlexCA_im2col<<<g1(tc), 256, 0, stream>>>(pool1, col, 64, 27, 27, 27, 27, 5, 1, 2, K, Kp, tc);
    unsigned long long tw = (unsigned long long)Kp * Np;
    AlexCA_packw<<<g1(tw), 256, 0, stream>>>(w2, Bm, N, K, Np, tw);
    AlexCA_gemm_core<false><<<dim3(Np / 64, cdiv(M, 128)), 256, 0, stream>>>(
        col, Bm, b2, act2, M, N, Kp, Np, Kp, 27 * 27, 1);
    unsigned long long tp = (unsigned long long)64 * 192 * 13 * 13;
    AlexCA_maxpool<<<g1(tp), 256, 0, stream>>>(act2, pool2, 64 * 192, 27, 27, 13, 13, tp);
  }
  // ---- conv3: 192x13x13 -> 384x13x13, k3 s1 p1, K=1728 ----
  {
    const int M = 64 * 13 * 13, K = 1728, Kp = 1728, N = 384, Np = 384;
    unsigned long long tc = (unsigned long long)M * Kp;
    AlexCA_im2col<<<g1(tc), 256, 0, stream>>>(pool2, col, 192, 13, 13, 13, 13, 3, 1, 1, K, Kp, tc);
    unsigned long long tw = (unsigned long long)Kp * Np;
    AlexCA_packw<<<g1(tw), 256, 0, stream>>>(w3, Bm, N, K, Np, tw);
    AlexCA_gemm_core<false><<<dim3(Np / 64, cdiv(M, 128)), 256, 0, stream>>>(
        col, Bm, b3, act3, M, N, Kp, Np, Kp, 13 * 13, 1);
  }
  // ---- conv4: 384x13x13 -> 256x13x13, K=3456 ----
  {
    const int M = 64 * 13 * 13, K = 3456, Kp = 3456, N = 256, Np = 256;
    unsigned long long tc = (unsigned long long)M * Kp;
    AlexCA_im2col<<<g1(tc), 256, 0, stream>>>(act3, col, 384, 13, 13, 13, 13, 3, 1, 1, K, Kp, tc);
    unsigned long long tw = (unsigned long long)Kp * Np;
    AlexCA_packw<<<g1(tw), 256, 0, stream>>>(w4, Bm, N, K, Np, tw);
    AlexCA_gemm_core<false><<<dim3(Np / 64, cdiv(M, 128)), 256, 0, stream>>>(
        col, Bm, b4, act4, M, N, Kp, Np, Kp, 13 * 13, 1);
  }
  // ---- conv5: 256x13x13 -> 256x13x13, K=2304 ----
  {
    const int M = 64 * 13 * 13, K = 2304, Kp = 2304, N = 256, Np = 256;
    unsigned long long tc = (unsigned long long)M * Kp;
    AlexCA_im2col<<<g1(tc), 256, 0, stream>>>(act4, col, 256, 13, 13, 13, 13, 3, 1, 1, K, Kp, tc);
    unsigned long long tw = (unsigned long long)Kp * Np;
    AlexCA_packw<<<g1(tw), 256, 0, stream>>>(w5, Bm, N, K, Np, tw);
    AlexCA_gemm_core<false><<<dim3(Np / 64, cdiv(M, 128)), 256, 0, stream>>>(
        col, Bm, b5, act5, M, N, Kp, Np, Kp, 13 * 13, 1);
    unsigned long long tp = (unsigned long long)64 * 256 * 6 * 6;
    AlexCA_maxpool<<<g1(tp), 256, 0, stream>>>(act5, pool3, 64 * 256, 13, 13, 6, 6, tp);
  }
  // ---- cmap + feature assembly ----
  AlexCA_stats<<<64, 256, 0, stream>>>(pool3, mv, qv);
  {
    unsigned long long ta = (unsigned long long)64 * 9216;
    AlexCA_featA<<<g1(ta), 256, 0, stream>>>(pool3, feat, ta);
    unsigned long long tb = (unsigned long long)64 * 65536;
    AlexCA_featB<<<g1(tb), 256, 0, stream>>>(mv, qv, feat, tb);
  }
  // ---- FC: [64,74752] @ [74752,1000] + bc, split-K for parallelism ----
  {
    const int M = 64, K = 74752, Kp = 74752, N = 1000, Np = 1024;
    unsigned long long tw = (unsigned long long)Kp * Np;
    AlexCA_packw<<<g1(tw), 256, 0, stream>>>(wc, Bm, N, K, Np, tw);
    AlexCA_gemm_core<true><<<dim3(Np / 64, cdiv(M, 128), NSPLIT), 256, 0, stream>>>(
        feat, Bm, nullptr, part, M, N, Kp, Np, KCH, 1, 0);
    unsigned long long tr = (unsigned long long)64 * 1000;
    AlexCA_fc_reduce<<<g1(tr), 256, 0, stream>>>(part, bc, out, NSPLIT, Np, tr);
  }
}
